// ThymeLM_71554155151770
// MI455X (gfx1250) — matrix operand order repaired and checked by reading, not compiled, not run on hardware
//
#include <hip/hip_runtime.h>
#include <hip/hip_bf16.h>
#include <math.h>

// ---------------- problem constants ----------------
#define B_  4
#define T_  1024
#define V_  32000
#define E_  256
#define NAX 12
#define NCP 24
#define NST 576          // 24*24
#define MTOK 4096        // B*T
#define H1  512          // hidden
#define K1  576          // GEMM1 K (= NST)
#define K2  512          // GEMM2 K (= H1)

typedef __attribute__((ext_vector_type(16))) __bf16 v16bf;
typedef __attribute__((ext_vector_type(8)))  float  v8f;
typedef __attribute__((ext_vector_type(4)))  float  nf4;     // native float4
typedef int v4i __attribute__((vector_size(16)));            // gcc-style int4

// -------- CDNA5 async global->LDS path (guarded; fallback = sync staging) ----
#if defined(__gfx1250__) && __has_builtin(__builtin_amdgcn_global_load_async_to_lds_b128)
#define USE_ASYNC 1
#else
#define USE_ASYNC 0
#endif

#if USE_ASYNC
__device__ __forceinline__ void async_cp16(const unsigned short* g, unsigned short* l) {
    __builtin_amdgcn_global_load_async_to_lds_b128(
        (__attribute__((address_space(1))) v4i*)g,
        (__attribute__((address_space(3))) v4i*)l,
        0, 0);
}
#if __has_builtin(__builtin_amdgcn_s_wait_asynccnt)
#define WAIT_ASYNC(n) __builtin_amdgcn_s_wait_asynccnt(n)
#else
#define WAIT_ASYNC(n) asm volatile("s_wait_asynccnt %0" :: "i"(n) : "memory")
#endif
#endif

// ---------------- helpers ----------------
__device__ __forceinline__ unsigned short f2bf(float f) {
    unsigned u = __builtin_bit_cast(unsigned, f);
    unsigned r = u + 0x7FFFu + ((u >> 16) & 1u);   // round-to-nearest-even
    return (unsigned short)(r >> 16);
}

__device__ __forceinline__ v16bf ld_frag2(const unsigned short* p0, const unsigned short* p1) {
    union { uint4 q[2]; v16bf v; } u;
    u.q[0] = *(const uint4*)p0;
    u.q[1] = *(const uint4*)p1;
    return u.v;
}

// =====================================================================
// Kernel 1: fp32 -> bf16 weight conversion (W1, W2)
// =====================================================================
#define N1C (H1 * K1)      // 294912
#define N2C (V_ * K2)      // 16384000
__global__ void convert_kernel(const float* __restrict__ W1, const float* __restrict__ W2,
                               unsigned short* __restrict__ W1b, unsigned short* __restrict__ W2b) {
    int idx = blockIdx.x * blockDim.x + threadIdx.x;
    long long i4 = (long long)idx * 4;
    union { unsigned short s[4]; uint2 v; } pk;
    if (i4 < N1C) {
        float4 v = *(const float4*)(W1 + i4);
        pk.s[0] = f2bf(v.x); pk.s[1] = f2bf(v.y); pk.s[2] = f2bf(v.z); pk.s[3] = f2bf(v.w);
        *(uint2*)(W1b + i4) = pk.v;
    } else if (i4 < (long long)N1C + N2C) {
        long long j = i4 - N1C;
        float4 v = *(const float4*)(W2 + j);
        pk.s[0] = f2bf(v.x); pk.s[1] = f2bf(v.y); pk.s[2] = f2bf(v.z); pk.s[3] = f2bf(v.w);
        *(uint2*)(W2b + j) = pk.v;
    }
}

// =====================================================================
// Kernel 2: front-end: embed gather -> tanh(LN(x W_ax^T + b)) -> tanh(LN(a W_comp^T + b))
// =====================================================================
__global__ void frontend_kernel(const int* __restrict__ ids, const float* __restrict__ emb,
                                const float* __restrict__ W_ax, const float* __restrict__ b_ax,
                                const float* __restrict__ g_ax, const float* __restrict__ beta_ax,
                                const float* __restrict__ W_comp, const float* __restrict__ b_comp,
                                const float* __restrict__ g_comp, const float* __restrict__ beta_comp,
                                float* __restrict__ c_buf) {
    __shared__ __align__(16) float sW[NAX * E_];
    __shared__ float sWc[NCP * NAX];
    __shared__ float sb[NAX], sg[NAX], sbe[NAX];
    __shared__ float sbc[NCP], sgc[NCP], sbec[NCP];

    int tid = threadIdx.x;
    for (int i = tid; i < NAX * E_; i += blockDim.x) sW[i] = W_ax[i];
    for (int i = tid; i < NCP * NAX; i += blockDim.x) sWc[i] = W_comp[i];
    if (tid < NAX) { sb[tid] = b_ax[tid]; sg[tid] = g_ax[tid]; sbe[tid] = beta_ax[tid]; }
    if (tid < NCP) { sbc[tid] = b_comp[tid]; sgc[tid] = g_comp[tid]; sbec[tid] = beta_comp[tid]; }
    __syncthreads();

    int tok = blockIdx.x * blockDim.x + tid;           // 0..4095
    int id = ids[tok];
    const float4* x4 = (const float4*)(emb + (long long)id * E_);
    const float4* w4 = (const float4*)sW;

    float s[NAX];
#pragma unroll
    for (int j = 0; j < NAX; ++j) s[j] = 0.f;
    for (int k = 0; k < E_ / 4; ++k) {
        float4 xv = x4[k];
#pragma unroll
        for (int j = 0; j < NAX; ++j) {
            float4 wv = w4[j * (E_ / 4) + k];
            s[j] += xv.x * wv.x + xv.y * wv.y + xv.z * wv.z + xv.w * wv.w;
        }
    }
    float mu = 0.f;
#pragma unroll
    for (int j = 0; j < NAX; ++j) { s[j] += sb[j]; mu += s[j]; }
    mu *= (1.f / NAX);
    float var = 0.f;
#pragma unroll
    for (int j = 0; j < NAX; ++j) { float d = s[j] - mu; var += d * d; }
    var *= (1.f / NAX);
    float rstd = rsqrtf(var + 1e-5f);
    float a[NAX];
#pragma unroll
    for (int j = 0; j < NAX; ++j) a[j] = tanhf((s[j] - mu) * rstd * sg[j] + sbe[j]);

    float p[NCP];
    float mu2 = 0.f;
#pragma unroll
    for (int i = 0; i < NCP; ++i) {
        float acc = sbc[i];
#pragma unroll
        for (int j = 0; j < NAX; ++j) acc += a[j] * sWc[i * NAX + j];
        p[i] = acc; mu2 += acc;
    }
    mu2 *= (1.f / NCP);
    float var2 = 0.f;
#pragma unroll
    for (int i = 0; i < NCP; ++i) { float d = p[i] - mu2; var2 += d * d; }
    var2 *= (1.f / NCP);
    float rstd2 = rsqrtf(var2 + 1e-5f);
#pragma unroll
    for (int i = 0; i < NCP; ++i)
        c_buf[(long long)tok * NCP + i] = tanhf((p[i] - mu2) * rstd2 * sgc[i] + sbec[i]);
}

// =====================================================================
// Kernel 3: decay scan -> flat^T (bf16, K-major) + final state (fp32)
// =====================================================================
__global__ void scan_kernel(const float* __restrict__ c_buf,
                            const float* __restrict__ decay_p, const float* __restrict__ mix_p,
                            unsigned short* __restrict__ flatT, float* __restrict__ out_final) {
    int b = blockIdx.x;
    int k = threadIdx.x;               // 0..575
    int i = k / NCP, j = k % NCP;
    float dd = 1.f / (1.f + expf(-decay_p[0]));
    float mm = 1.f / (1.f + expf(-mix_p[0]));
    const float* cb = c_buf + (long long)b * T_ * NCP;
    unsigned short* dst = flatT + (long long)k * MTOK + b * T_;
    float s = 0.f;
    for (int t8 = 0; t8 < T_ / 8; ++t8) {
        union { unsigned short u[8]; uint4 v; } pk;
#pragma unroll
        for (int e = 0; e < 8; ++e) {
            int t = t8 * 8 + e;
            float ci = cb[t * NCP + i];
            float cj = cb[t * NCP + j];
            s = dd * s + mm * ci * cj;
            pk.u[e] = f2bf(s);
        }
        *(uint4*)(dst + t8 * 8) = pk.v;
    }
    out_final[(long long)b * NST + k] = s;
}

// =====================================================================
// GEMM core: D'[M=weight-out][N=tokens] = A(W, k-contig) x B(act^T, K-major)
// 128x128 tile, 8 waves, 4x2 v_wmma_f32_16x16x32_bf16 per wave.
// Async-DMA double-buffered LDS staging on CDNA5.
// =====================================================================
#define LDSA 40    // 32 + 8 bf16 pad  (80B row stride, 16B aligned, conflict-free)
#define LDSB 136   // 128 + 8 bf16 pad (272B row stride)

// ---- GEMM1: hT[j][m] = gelu( sum_k W1[j][k]*flatT[k][m] + b1[j] ), bf16 out
__global__ __launch_bounds__(256) void gemm1_kernel(const unsigned short* __restrict__ A,
                                                    const unsigned short* __restrict__ Bm,
                                                    const float* __restrict__ b1,
                                                    unsigned short* __restrict__ hT) {
    __shared__ __align__(16) unsigned short As[2][128 * LDSA];
    __shared__ __align__(16) unsigned short Bs[2][32 * LDSB];
    __shared__ float sBias[128];

    int tid = threadIdx.x;
    int wid = tid >> 5, lane = tid & 31, half = lane >> 4, r = lane & 15;
    int waveM = wid & 1, waveN = wid >> 1;
    int mTile = blockIdx.y * 128;     // over j (H1)
    int nTile = blockIdx.x * 128;     // over tokens

    if (tid < 128) sBias[tid] = b1[mTile + tid];

    int rowA = tid >> 1, koffA = (tid & 1) * 16;
    int rowB = tid >> 3, moffB = (tid & 7) * 16;
    const unsigned short* gA0 = A  + (long long)(mTile + rowA) * K1 + koffA;
    const unsigned short* gB0 = Bm + (long long)rowB * MTOK + nTile + moffB;

    v8f acc[4][2];
#pragma unroll
    for (int s4 = 0; s4 < 4; ++s4)
#pragma unroll
        for (int u = 0; u < 2; ++u) acc[s4][u] = (v8f)(0.f);

    const int KT = K1 / 32;   // 18

#if USE_ASYNC
    {
        const unsigned short* ga = gA0;
        unsigned short* la = &As[0][rowA * LDSA + koffA];
        async_cp16(ga, la); async_cp16(ga + 8, la + 8);
        const unsigned short* gb = gB0;
        unsigned short* lb = &Bs[0][rowB * LDSB + moffB];
        async_cp16(gb, lb); async_cp16(gb + 8, lb + 8);
    }
    for (int kt = 0; kt < KT; ++kt) {
        int cur = kt & 1;
        if (kt + 1 < KT) {
            const unsigned short* ga = gA0 + (kt + 1) * 32;
            unsigned short* la = &As[cur ^ 1][rowA * LDSA + koffA];
            async_cp16(ga, la); async_cp16(ga + 8, la + 8);
            const unsigned short* gb = gB0 + (long long)(kt + 1) * 32 * MTOK;
            unsigned short* lb = &Bs[cur ^ 1][rowB * LDSB + moffB];
            async_cp16(gb, lb); async_cp16(gb + 8, lb + 8);
            WAIT_ASYNC(4);
        } else {
            WAIT_ASYNC(0);
        }
        if (kt + 2 < KT) {
            __builtin_prefetch(gA0 + (kt + 2) * 32, 0, 1);
            __builtin_prefetch(gB0 + (long long)(kt + 2) * 32 * MTOK, 0, 1);
        }
        __syncthreads();

        v16bf af[4], bfr[2];
#pragma unroll
        for (int s4 = 0; s4 < 4; ++s4) {
            int base = (waveM * 64 + s4 * 16 + r) * LDSA + half * 8;
            af[s4] = ld_frag2(&As[cur][base], &As[cur][base + 16]);
        }
#pragma unroll
        for (int u = 0; u < 2; ++u) {
            int eb = waveN * 32 + u * 16 + r;
            union { unsigned short e[16]; v16bf v; } tb;
#pragma unroll
            for (int e = 0; e < 16; ++e) tb.e[e] = Bs[cur][((e < 8) ? (half * 8 + e) : (16 + half * 8 + e - 8)) * LDSB + eb];
            bfr[u] = tb.v;
        }
#pragma unroll
        for (int s4 = 0; s4 < 4; ++s4)
#pragma unroll
            for (int u = 0; u < 2; ++u)
                acc[s4][u] = __builtin_amdgcn_wmma_f32_16x16x32_bf16(
                    false, af[s4], false, bfr[u], (short)0, acc[s4][u], false, false);
        __syncthreads();
    }
#else
    for (int kt = 0; kt < KT; ++kt) {
        {
            const uint4* g = (const uint4*)(gA0 + kt * 32);
            uint4 d0 = g[0], d1 = g[1];
            uint4* d = (uint4*)(&As[0][rowA * LDSA + koffA]);
            d[0] = d0; d[1] = d1;
        }
        {
            const uint4* g = (const uint4*)(gB0 + (long long)kt * 32 * MTOK);
            uint4 d0 = g[0], d1 = g[1];
            uint4* d = (uint4*)(&Bs[0][rowB * LDSB + moffB]);
            d[0] = d0; d[1] = d1;
        }
        if (kt + 1 < KT) {
            __builtin_prefetch(gA0 + (kt + 1) * 32, 0, 1);
            __builtin_prefetch(gB0 + (long long)(kt + 1) * 32 * MTOK, 0, 1);
        }
        __syncthreads();
        v16bf af[4], bfr[2];
#pragma unroll
        for (int s4 = 0; s4 < 4; ++s4) {
            int base = (waveM * 64 + s4 * 16 + r) * LDSA + half * 8;
            af[s4] = ld_frag2(&As[0][base], &As[0][base + 16]);
        }
#pragma unroll
        for (int u = 0; u < 2; ++u) {
            int eb = waveN * 32 + u * 16 + r;
            union { unsigned short e[16]; v16bf v; } tb;
#pragma unroll
            for (int e = 0; e < 16; ++e) tb.e[e] = Bs[0][((e < 8) ? (half * 8 + e) : (16 + half * 8 + e - 8)) * LDSB + eb];
            bfr[u] = tb.v;
        }
#pragma unroll
        for (int s4 = 0; s4 < 4; ++s4)
#pragma unroll
            for (int u = 0; u < 2; ++u)
                acc[s4][u] = __builtin_amdgcn_wmma_f32_16x16x32_bf16(
                    false, af[s4], false, bfr[u], (short)0, acc[s4][u], false, false);
        __syncthreads();
    }
#endif

    // epilogue: + b1, exact GELU, store bf16 to hT[j][m]
#pragma unroll
    for (int s4 = 0; s4 < 4; ++s4) {
        int j0 = waveM * 64 + s4 * 16 + 8 * half;   // local j
#pragma unroll
        for (int u = 0; u < 2; ++u) {
            int m = nTile + waveN * 32 + u * 16 + r;
#pragma unroll
            for (int g = 0; g < 8; ++g) {
                int jl = j0 + g;
                float x = acc[s4][u][g] + sBias[jl];
                float h = 0.5f * x * (1.f + erff(x * 0.70710678118654752f));
                hT[(long long)(mTile + jl) * MTOK + m] = f2bf(h);
            }
        }
    }
}

// ---- GEMM2: logits[m][n] = sum_k W2[n][k]*hT[k][m] + b2[n], fp32 NT-stores
__global__ __launch_bounds__(256) void gemm2_kernel(const unsigned short* __restrict__ A,
                                                    const unsigned short* __restrict__ Bm,
                                                    const float* __restrict__ b2,
                                                    float* __restrict__ logits) {
    __shared__ __align__(16) unsigned short As[2][128 * LDSA];
    __shared__ __align__(16) unsigned short Bs[2][32 * LDSB];
    __shared__ float sBias[128];

    int tid = threadIdx.x;
    int wid = tid >> 5, lane = tid & 31, half = lane >> 4, r = lane & 15;
    int waveM = wid & 1, waveN = wid >> 1;
    int nTile = blockIdx.y * 128;     // over vocab
    int mTile = blockIdx.x * 128;     // over tokens

    if (tid < 128) sBias[tid] = b2[nTile + tid];

    int rowA = tid >> 1, koffA = (tid & 1) * 16;
    int rowB = tid >> 3, moffB = (tid & 7) * 16;
    const unsigned short* gA0 = A  + (long long)(nTile + rowA) * K2 + koffA;
    const unsigned short* gB0 = Bm + (long long)rowB * MTOK + mTile + moffB;

    v8f acc[4][2];
#pragma unroll
    for (int s4 = 0; s4 < 4; ++s4)
#pragma unroll
        for (int u = 0; u < 2; ++u) acc[s4][u] = (v8f)(0.f);

    const int KT = K2 / 32;   // 16

#if USE_ASYNC
    {
        const unsigned short* ga = gA0;
        unsigned short* la = &As[0][rowA * LDSA + koffA];
        async_cp16(ga, la); async_cp16(ga + 8, la + 8);
        const unsigned short* gb = gB0;
        unsigned short* lb = &Bs[0][rowB * LDSB + moffB];
        async_cp16(gb, lb); async_cp16(gb + 8, lb + 8);
    }
    for (int kt = 0; kt < KT; ++kt) {
        int cur = kt & 1;
        if (kt + 1 < KT) {
            const unsigned short* ga = gA0 + (kt + 1) * 32;
            unsigned short* la = &As[cur ^ 1][rowA * LDSA + koffA];
            async_cp16(ga, la); async_cp16(ga + 8, la + 8);
            const unsigned short* gb = gB0 + (long long)(kt + 1) * 32 * MTOK;
            unsigned short* lb = &Bs[cur ^ 1][rowB * LDSB + moffB];
            async_cp16(gb, lb); async_cp16(gb + 8, lb + 8);
            WAIT_ASYNC(4);
        } else {
            WAIT_ASYNC(0);
        }
        if (kt + 2 < KT) {
            __builtin_prefetch(gA0 + (kt + 2) * 32, 0, 1);
            __builtin_prefetch(gB0 + (long long)(kt + 2) * 32 * MTOK, 0, 1);
        }
        __syncthreads();

        v16bf af[4], bfr[2];
#pragma unroll
        for (int s4 = 0; s4 < 4; ++s4) {
            int base = (waveM * 64 + s4 * 16 + r) * LDSA + half * 8;
            af[s4] = ld_frag2(&As[cur][base], &As[cur][base + 16]);
        }
#pragma unroll
        for (int u = 0; u < 2; ++u) {
            int eb = waveN * 32 + u * 16 + r;
            union { unsigned short e[16]; v16bf v; } tb;
#pragma unroll
            for (int e = 0; e < 16; ++e) tb.e[e] = Bs[cur][((e < 8) ? (half * 8 + e) : (16 + half * 8 + e - 8)) * LDSB + eb];
            bfr[u] = tb.v;
        }
#pragma unroll
        for (int s4 = 0; s4 < 4; ++s4)
#pragma unroll
            for (int u = 0; u < 2; ++u)
                acc[s4][u] = __builtin_amdgcn_wmma_f32_16x16x32_bf16(
                    false, af[s4], false, bfr[u], (short)0, acc[s4][u], false, false);
        __syncthreads();
    }
#else
    for (int kt = 0; kt < KT; ++kt) {
        {
            const uint4* g = (const uint4*)(gA0 + kt * 32);
            uint4 d0 = g[0], d1 = g[1];
            uint4* d = (uint4*)(&As[0][rowA * LDSA + koffA]);
            d[0] = d0; d[1] = d1;
        }
        {
            const uint4* g = (const uint4*)(gB0 + (long long)kt * 32 * MTOK);
            uint4 d0 = g[0], d1 = g[1];
            uint4* d = (uint4*)(&Bs[0][rowB * LDSB + moffB]);
            d[0] = d0; d[1] = d1;
        }
        if (kt + 1 < KT) {
            __builtin_prefetch(gA0 + (kt + 1) * 32, 0, 1);
            __builtin_prefetch(gB0 + (long long)(kt + 1) * 32 * MTOK, 0, 1);
        }
        __syncthreads();
        v16bf af[4], bfr[2];
#pragma unroll
        for (int s4 = 0; s4 < 4; ++s4) {
            int base = (waveM * 64 + s4 * 16 + r) * LDSA + half * 8;
            af[s4] = ld_frag2(&As[0][base], &As[0][base + 16]);
        }
#pragma unroll
        for (int u = 0; u < 2; ++u) {
            int eb = waveN * 32 + u * 16 + r;
            union { unsigned short e[16]; v16bf v; } tb;
#pragma unroll
            for (int e = 0; e < 16; ++e) tb.e[e] = Bs[0][((e < 8) ? (half * 8 + e) : (16 + half * 8 + e - 8)) * LDSB + eb];
            bfr[u] = tb.v;
        }
#pragma unroll
        for (int s4 = 0; s4 < 4; ++s4)
#pragma unroll
            for (int u = 0; u < 2; ++u)
                acc[s4][u] = __builtin_amdgcn_wmma_f32_16x16x32_bf16(
                    false, af[s4], false, bfr[u], (short)0, acc[s4][u], false, false);
        __syncthreads();
    }
#endif

    // epilogue: un-transpose for free; non-temporal native-float4 stores (524MB
    // stream must not evict the L2-resident W2/hT read set).
#pragma unroll
    for (int s4 = 0; s4 < 4; ++s4) {
        int n0l = waveM * 64 + s4 * 16 + 8 * half;    // local n
        int n0 = nTile + n0l;
#pragma unroll
        for (int u = 0; u < 2; ++u) {
            int m = mTile + waveN * 32 + u * 16 + r;
            float* outp = logits + (long long)m * V_ + n0;
            nf4 v0, v1;
            v0.x = acc[s4][u][0] + sBias[n0l + 0];
            v0.y = acc[s4][u][1] + sBias[n0l + 1];
            v0.z = acc[s4][u][2] + sBias[n0l + 2];
            v0.w = acc[s4][u][3] + sBias[n0l + 3];
            v1.x = acc[s4][u][4] + sBias[n0l + 4];
            v1.y = acc[s4][u][5] + sBias[n0l + 5];
            v1.z = acc[s4][u][6] + sBias[n0l + 6];
            v1.w = acc[s4][u][7] + sBias[n0l + 7];
            __builtin_nontemporal_store(v0, (nf4*)(outp));
            __builtin_nontemporal_store(v1, (nf4*)(outp + 4));
        }
    }
}

// =====================================================================
// launch
// =====================================================================
static inline size_t align256(size_t x) { return (x + 255) & ~(size_t)255; }

extern "C" void kernel_launch(void* const* d_in, const int* in_sizes, int n_in,
                              void* d_out, int out_size, void* d_ws, size_t ws_size,
                              hipStream_t stream) {
    const int*   input_ids = (const int*)d_in[0];
    const float* emb       = (const float*)d_in[1];
    const float* W_ax      = (const float*)d_in[2];
    const float* b_ax      = (const float*)d_in[3];
    const float* g_ax      = (const float*)d_in[4];
    const float* beta_ax   = (const float*)d_in[5];
    const float* W_comp    = (const float*)d_in[6];
    const float* b_comp    = (const float*)d_in[7];
    const float* g_comp    = (const float*)d_in[8];
    const float* beta_comp = (const float*)d_in[9];
    const float* decay_p   = (const float*)d_in[10];
    const float* mix_p     = (const float*)d_in[11];
    const float* W1        = (const float*)d_in[12];
    const float* b1        = (const float*)d_in[13];
    const float* W2        = (const float*)d_in[14];
    const float* b2        = (const float*)d_in[15];

    float* logits = (float*)d_out;                               // [4096][32000]
    float* final_state = logits + (long long)MTOK * V_;          // [B][24][24]

    // workspace carve-up (~43 MB)
    char* ws = (char*)d_ws;
    size_t off = 0;
    float* c_buf = (float*)(ws + off);                   off = align256(off + (size_t)MTOK * NCP * 4);
    unsigned short* flatT = (unsigned short*)(ws + off); off = align256(off + (size_t)K1 * MTOK * 2);
    unsigned short* hT    = (unsigned short*)(ws + off); off = align256(off + (size_t)K2 * MTOK * 2);
    unsigned short* W1b   = (unsigned short*)(ws + off); off = align256(off + (size_t)N1C * 2);
    unsigned short* W2b   = (unsigned short*)(ws + off); off = align256(off + (size_t)N2C * 2);

    // 1) weight conversion fp32 -> bf16
    {
        long long total4 = ((long long)N1C + N2C) / 4;
        int blocks = (int)((total4 + 255) / 256);
        convert_kernel<<<blocks, 256, 0, stream>>>(W1, W2, W1b, W2b);
    }
    // 2) front-end (one thread per token)
    frontend_kernel<<<MTOK / 128, 128, 0, stream>>>(input_ids, emb, W_ax, b_ax, g_ax, beta_ax,
                                                    W_comp, b_comp, g_comp, beta_comp, c_buf);
    // 3) decay scan -> flat^T (bf16) + final state (fp32)
    scan_kernel<<<B_, NST, 0, stream>>>(c_buf, decay_p, mix_p, flatT, final_state);
    // 4) GEMM1 -> hT (bf16), fused bias+GELU
    gemm1_kernel<<<dim3(MTOK / 128, H1 / 128), 256, 0, stream>>>(W1b, flatT, b1, hT);
    // 5) GEMM2 -> logits (fp32), fused bias
    gemm2_kernel<<<dim3(MTOK / 128, V_ / 128), 256, 0, stream>>>(W2b, hT, b2, logits);
}
